// Block_48447231099081
// MI455X (gfx1250) — compile-verified, matrix operand order checked by reading, not run on hardware
//
#include <hip/hip_runtime.h>

// ---------------------------------------------------------------------------
// Types for CDNA5 WMMA (wave32): v_wmma_f32_16x16x32_bf16
// ---------------------------------------------------------------------------
typedef __bf16 v16bf __attribute__((ext_vector_type(16)));
typedef __bf16 v8bf  __attribute__((ext_vector_type(8)));
typedef float  v8f   __attribute__((ext_vector_type(8)));

__device__ __forceinline__ v8f wmma_bf16(v16bf a, v16bf b, v8f c) {
  // (neg_a, A, neg_b, B, c_mod, C, reuse_a, reuse_b)
  return __builtin_amdgcn_wmma_f32_16x16x32_bf16(false, a, false, b, (short)0, c,
                                                 false, false);
}

// Async global->LDS copy, 16 bytes per lane (CDNA5, ASYNCcnt-tracked).
__device__ __forceinline__ void async_b128(void* lds_dst, const void* gsrc) {
  unsigned l = (unsigned)(uintptr_t)lds_dst;  // generic LDS ptr: low 32b = LDS offset
  asm volatile("global_load_async_to_lds_b128 %0, %1, off"
               :: "v"(l), "v"(gsrc) : "memory");
}
#define WAIT_ASYNC(n) asm volatile("s_wait_asynccnt %0" :: "i"(n) : "memory")

// Load a 16-element bf16 fragment for one lane.
// A (row m = lane&15): elems {k0+h*8..+7} and {k0+h*8+16..+7}, h = lane>>4.
// B stored transposed [N][K]: identical pattern with n = lane&15.
// `p` points at row_start + k0 + (lane>>4)*8 and is 16-byte aligned.
__device__ __forceinline__ v16bf load_frag(const __bf16* p) {
  v8bf lo = *(const v8bf*)(p);
  v8bf hi = *(const v8bf*)(p + 16);
  v16bf f;
#pragma unroll
  for (int i = 0; i < 8; ++i) { f[i] = lo[i]; f[8 + i] = hi[i]; }
  return f;
}

// ---------------------------------------------------------------------------
// Problem constants
// ---------------------------------------------------------------------------
#define BB   2
#define SEQ  2048
#define DIM  2048
#define NH   16
#define HDIM 128
#define DFF  8192
#define BS   (BB * SEQ)   // 4096 rows

// ---------------------------------------------------------------------------
// Weight convert + transpose via 32x32 LDS tile: coalesced both sides.
// wT[n*K + k] = bf16(w[k*N + n]);  grid = (N/32, K/32), 256 threads.
// ---------------------------------------------------------------------------
__global__ __launch_bounds__(256)
void cvt_transpose_kernel(const float* __restrict__ w, __bf16* __restrict__ wT,
                          int K, int N) {
  __shared__ float t[32][33];
  const int kb = blockIdx.y * 32, nb = blockIdx.x * 32;
  const int lx = threadIdx.x & 31, ly = threadIdx.x >> 5;  // 32 x 8
#pragma unroll
  for (int dy = 0; dy < 32; dy += 8)
    t[ly + dy][lx] = w[(size_t)(kb + ly + dy) * N + nb + lx];
  __syncthreads();
#pragma unroll
  for (int dy = 0; dy < 32; dy += 8)
    wT[(size_t)(nb + ly + dy) * K + kb + lx] = (__bf16)t[lx][ly + dy];
}

// ---------------------------------------------------------------------------
// LayerNorm: one 256-thread block per row; outputs f32 and bf16 copies
// ---------------------------------------------------------------------------
__device__ __forceinline__ float block_reduce_sum(float v) {
#pragma unroll
  for (int m = 16; m >= 1; m >>= 1) v += __shfl_xor(v, m, 32);
  __shared__ float tmp[8];
  __shared__ float total;
  int lane = threadIdx.x & 31, w = threadIdx.x >> 5;
  if (lane == 0) tmp[w] = v;
  __syncthreads();
  float r = (threadIdx.x < 8) ? tmp[threadIdx.x] : 0.f;
  if (w == 0) {
#pragma unroll
    for (int m = 4; m >= 1; m >>= 1) r += __shfl_xor(r, m, 32);
  }
  if (threadIdx.x == 0) total = r;
  __syncthreads();
  return total;
}

__global__ __launch_bounds__(256)
void layernorm_kernel(const float* __restrict__ x, const float* __restrict__ sc,
                      const float* __restrict__ bi, float* __restrict__ outF,
                      __bf16* __restrict__ outB, int D) {
  const size_t row = blockIdx.x;
  const float* xr = x + row * (size_t)D;
  float s = 0.f;
  for (int i = threadIdx.x; i < D; i += blockDim.x) s += xr[i];
  float mu = block_reduce_sum(s) / (float)D;
  float v = 0.f;
  for (int i = threadIdx.x; i < D; i += blockDim.x) {
    float d = xr[i] - mu; v += d * d;
  }
  float rstd = rsqrtf(block_reduce_sum(v) / (float)D + 1e-6f);
  for (int i = threadIdx.x; i < D; i += blockDim.x) {
    float h = (xr[i] - mu) * rstd * sc[i] + bi[i];
    outF[row * (size_t)D + i] = h;
    outB[row * (size_t)D + i] = (__bf16)h;
  }
}

// ---------------------------------------------------------------------------
// bf16 WMMA GEMM: C[M,N] = A[M,K] * B[K,N]  (B supplied transposed: BT[N][K])
// Block tile 128x128, K-step 32, 256 threads = 8 waves, each wave 32x64.
// Double-buffered LDS; tiles staged with async global->LDS copies.
// Epilogue: optional bias[n], ReLU, f32 residual, f32 and/or bf16 store.
// M,N % 128 == 0 ; K % 32 == 0.
// ---------------------------------------------------------------------------
#define GBM 128
#define GBN 128
#define GBK 32
#define GPAD 8   // 40-element rows: 80B stride, 16B-aligned, conflict-padded

__device__ __forceinline__ void stage_tile_async(__bf16 (*dst)[GBK + GPAD],
                                                 const __bf16* src, int ld,
                                                 int tid) {
  // 128 rows x 32 cols bf16 = 512 x 16B chunks; 256 threads -> 2 each.
#pragma unroll
  for (int t = 0; t < 2; ++t) {
    int i = tid + t * 256;
    int r = i >> 2, c = (i & 3) * 8;
    async_b128(&dst[r][c], src + (size_t)r * ld + c);
  }
}

__global__ __launch_bounds__(256)
void gemm_bf16_kernel(const __bf16* __restrict__ A, const __bf16* __restrict__ BT,
                      int M, int N, int K,
                      const float* __restrict__ bias,
                      const float* __restrict__ residual, int relu,
                      float* __restrict__ outF, __bf16* __restrict__ outB) {
  __shared__ __align__(16) __bf16 As[2][GBM][GBK + GPAD];
  __shared__ __align__(16) __bf16 Bs[2][GBN][GBK + GPAD];

  const int tid = threadIdx.x;
  const int wave = tid >> 5, lane = tid & 31;
  const int half = lane >> 4, l16 = lane & 15;
  const int bm = blockIdx.y * GBM, bn = blockIdx.x * GBN;
  const int wm = wave & 3;   // 4 row-groups of 32
  const int wn = wave >> 2;  // 2 col-groups of 64
  const __bf16* Ab = A + (size_t)bm * K;
  const __bf16* Bb = BT + (size_t)bn * K;

  v8f acc[2][4] = {};

  // Prologue: stage k-block 0 into buffer 0.
  stage_tile_async(As[0], Ab, K, tid);
  stage_tile_async(Bs[0], Bb, K, tid);

  int buf = 0;
  for (int kb = 0; kb < K; kb += GBK) {
    if (kb + GBK < K) {
      // Prefetch next k-block into the other buffer (4 async ops/thread),
      // then retire everything older (in-order completion).
      stage_tile_async(As[buf ^ 1], Ab + kb + GBK, K, tid);
      stage_tile_async(Bs[buf ^ 1], Bb + kb + GBK, K, tid);
      WAIT_ASYNC(4);
    } else {
      WAIT_ASYNC(0);
    }
    __syncthreads();

    v16bf af[2], bf[4];
#pragma unroll
    for (int i = 0; i < 2; ++i)
      af[i] = load_frag(&As[buf][wm * 32 + i * 16 + l16][half * 8]);
#pragma unroll
    for (int j = 0; j < 4; ++j)
      bf[j] = load_frag(&Bs[buf][wn * 64 + j * 16 + l16][half * 8]);

#pragma unroll
    for (int i = 0; i < 2; ++i)
#pragma unroll
      for (int j = 0; j < 4; ++j)
        acc[i][j] = wmma_bf16(af[i], bf[j], acc[i][j]);

    __syncthreads();  // everyone done reading `buf` before it is re-staged
    buf ^= 1;
  }

  // Epilogue. D layout: reg r -> row (r + 8*half), col = l16 (within 16x16).
#pragma unroll
  for (int i = 0; i < 2; ++i) {
#pragma unroll
    for (int j = 0; j < 4; ++j) {
      const int col = bn + wn * 64 + j * 16 + l16;
      const float bv = bias ? bias[col] : 0.f;
#pragma unroll
      for (int r = 0; r < 8; ++r) {
        const int row = bm + wm * 32 + i * 16 + r + half * 8;
        float v = acc[i][j][r] + bv;
        if (relu) v = fmaxf(v, 0.f);
        if (residual) v += residual[(size_t)row * N + col];
        if (outF) outF[(size_t)row * N + col] = v;
        if (outB) outB[(size_t)row * N + col] = (__bf16)v;
      }
    }
  }
}

// ---------------------------------------------------------------------------
// Flash attention (causal), bf16 in/out, f32 online softmax.
// Grid: (SEQ/64, NH, BB), 128 threads = 4 waves; wave w owns query rows w*16..+15.
// Keys/values in blocks of 32. Q/K/V/O layout: [b, s, h, hd] (h,hd packed).
// Q and K tiles staged with async global->LDS; V transposed via LDS.
// ---------------------------------------------------------------------------
__global__ __launch_bounds__(128)
void flash_attn_kernel(const __bf16* __restrict__ Q, const __bf16* __restrict__ Km,
                       const __bf16* __restrict__ V, __bf16* __restrict__ O) {
  __shared__ __align__(16) __bf16 Qs[64][HDIM + 8];
  __shared__ __align__(16) __bf16 Ks[32][HDIM + 8];
  __shared__ __align__(16) __bf16 VT[HDIM][32 + 8];
  __shared__ __align__(16) __bf16 Ps[4][16][32 + 8];

  const int qt = blockIdx.x, h = blockIdx.y, b = blockIdx.z;
  const int D = NH * HDIM;
  const int tid = threadIdx.x, wave = tid >> 5, lane = tid & 31;
  const int half = lane >> 4, l16 = lane & 15;
  const size_t base = (size_t)b * SEQ * D + (size_t)h * HDIM;

  // Stage the 64x128 Q tile once (async).
#pragma unroll
  for (int t = 0; t < 8; ++t) {
    int i = tid + t * 128;
    int r = i >> 4, c = (i & 15) * 8;
    async_b128(&Qs[r][c], &Q[base + (size_t)(qt * 64 + r) * D + c]);
  }
  WAIT_ASYNC(0);
  __syncthreads();

  v8f o_acc[8] = {};            // 16 rows x 128 cols of O, per wave
  float m_r[8], l_r[8];
#pragma unroll
  for (int r = 0; r < 8; ++r) { m_r[r] = -1e30f; l_r[r] = 0.f; }

  const float scale = 0.08838834764831845f;  // HD^-0.5
  const int nkb = qt * 2 + 2;                // causal: keys up to qt*64+63

  for (int kb = 0; kb < nkb; ++kb) {
    // K tile (32x128) row-major == BT for S = Q*K^T: async copy.
#pragma unroll
    for (int t = 0; t < 4; ++t) {
      int i = tid + t * 128;
      int r = i >> 4, c = (i & 15) * 8;
      async_b128(&Ks[r][c], &Km[base + (size_t)(kb * 32 + r) * D + c]);
    }
    // V transposed into VT[hd][key]: vector global load, scalar LDS scatter.
#pragma unroll
    for (int t = 0; t < 4; ++t) {
      int i = tid + t * 128;
      int r = i >> 4, c0 = (i & 15) * 8;
      v8bf vv = *(const v8bf*)&V[base + (size_t)(kb * 32 + r) * D + c0];
#pragma unroll
      for (int j = 0; j < 8; ++j) VT[c0 + j][r] = vv[j];
    }
    WAIT_ASYNC(0);
    __syncthreads();

    // Scores: 16 queries x 32 keys, K-dim = 128 (4 WMMA per 16-key tile).
    v8f sacc[2] = {};
#pragma unroll
    for (int nt = 0; nt < 2; ++nt)
#pragma unroll
      for (int k0 = 0; k0 < HDIM; k0 += 32) {
        v16bf a = load_frag(&Qs[wave * 16 + l16][k0 + half * 8]);
        v16bf bb = load_frag(&Ks[nt * 16 + l16][k0 + half * 8]);
        sacc[nt] = wmma_bf16(a, bb, sacc[nt]);
      }

    // Mask + scale + online softmax update.
    float mnew[8];
#pragma unroll
    for (int r = 0; r < 8; ++r) {
      const int qg = qt * 64 + wave * 16 + r + half * 8;
      float mx = -1e30f;
#pragma unroll
      for (int nt = 0; nt < 2; ++nt) {
        const int kg = kb * 32 + nt * 16 + l16;
        float s = sacc[nt][r] * scale;
        if (kg > qg) s = -1e30f;
        sacc[nt][r] = s;
        mx = fmaxf(mx, s);
      }
#pragma unroll
      for (int m = 8; m >= 1; m >>= 1) mx = fmaxf(mx, __shfl_xor(mx, m, 32));
      mnew[r] = fmaxf(m_r[r], mx);
    }
#pragma unroll
    for (int r = 0; r < 8; ++r) {
      float rs = 0.f;
#pragma unroll
      for (int nt = 0; nt < 2; ++nt) {
        float p = __expf(sacc[nt][r] - mnew[r]);
        rs += p;
        Ps[wave][r + half * 8][nt * 16 + l16] = (__bf16)p;
      }
#pragma unroll
      for (int m = 8; m >= 1; m >>= 1) rs += __shfl_xor(rs, m, 32);
      const float corr = __expf(m_r[r] - mnew[r]);
      l_r[r] = l_r[r] * corr + rs;
      m_r[r] = mnew[r];
#pragma unroll
      for (int n = 0; n < 8; ++n) o_acc[n][r] *= corr;
    }
    // Same-wave LDS RAW on Ps before re-reading it as an A fragment.
    asm volatile("s_wait_dscnt 0x0" ::: "memory");

    // O += P (16x32) * V (32x128): 8 WMMA.
    v16bf pa = load_frag(&Ps[wave][l16][half * 8]);
#pragma unroll
    for (int n = 0; n < 8; ++n) {
      v16bf vb = load_frag(&VT[n * 16 + l16][half * 8]);
      o_acc[n] = wmma_bf16(pa, vb, o_acc[n]);
    }
    __syncthreads();
  }

  // Normalize and store bf16.
#pragma unroll
  for (int n = 0; n < 8; ++n) {
#pragma unroll
    for (int r = 0; r < 8; ++r) {
      const int qg = qt * 64 + wave * 16 + r + half * 8;
      const int col = n * 16 + l16;
      O[base + (size_t)qg * D + col] = (__bf16)(o_acc[n][r] / l_r[r]);
    }
  }
}

// ---------------------------------------------------------------------------
// Host orchestration
// ---------------------------------------------------------------------------
extern "C" void kernel_launch(void* const* d_in, const int* in_sizes, int n_in,
                              void* d_out, int out_size, void* d_ws, size_t ws_size,
                              hipStream_t stream) {
  (void)in_sizes; (void)n_in; (void)out_size; (void)ws_size;

  const float* x    = (const float*)d_in[0];
  const float* ln1s = (const float*)d_in[1];
  const float* ln1b = (const float*)d_in[2];
  const float* wq   = (const float*)d_in[3];
  const float* wk   = (const float*)d_in[4];
  const float* wv   = (const float*)d_in[5];
  const float* wo1  = (const float*)d_in[6];
  const float* wo2  = (const float*)d_in[7];
  const float* ln2s = (const float*)d_in[8];
  const float* ln2b = (const float*)d_in[9];
  const float* fw1  = (const float*)d_in[10];
  const float* fb1  = (const float*)d_in[11];
  const float* fw2  = (const float*)d_in[12];
  const float* fb2  = (const float*)d_in[13];
  float* out = (float*)d_out;

  // Workspace carve-up (256B aligned).
  char* ws = (char*)d_ws;
  size_t off = 0;
  auto alloc = [&](size_t bytes) -> char* {
    char* p = ws + off;
    off += (bytes + 255) & ~(size_t)255;
    return p;
  };
  const size_t DD  = (size_t)DIM * DIM;
  const size_t DDF = (size_t)DIM * DFF;
  const size_t AD  = (size_t)BS * DIM;
  const size_t ADF = (size_t)BS * DFF;

  __bf16* wqT  = (__bf16*)alloc(DD * 2);
  __bf16* wkT  = (__bf16*)alloc(DD * 2);
  __bf16* wvT  = (__bf16*)alloc(DD * 2);
  __bf16* wo1T = (__bf16*)alloc(DD * 2);
  __bf16* wo2T = (__bf16*)alloc(DD * 2);
  __bf16* fw1T = (__bf16*)alloc(DDF * 2);
  __bf16* fw2T = (__bf16*)alloc(DDF * 2);
  float*  h_f  = (float*)alloc(AD * 4);   // later reused as h2_f
  __bf16* h_b  = (__bf16*)alloc(AD * 2);
  __bf16* q_b  = (__bf16*)alloc(AD * 2);  // later reused as o1_b
  __bf16* k_b  = (__bf16*)alloc(AD * 2);  // later reused as h2_b
  __bf16* v_b  = (__bf16*)alloc(AD * 2);
  __bf16* at_b = (__bf16*)alloc(AD * 2);
  float*  x1_f = (float*)alloc(AD * 4);
  __bf16* f1_b = (__bf16*)alloc(ADF * 2);
  __bf16* o1_b = q_b;
  float*  h2_f = h_f;
  __bf16* h2_b = k_b;

  // 1) Weights: convert + transpose to bf16 [N][K] (coalesced LDS transpose).
  dim3 tDD(DIM / 32, DIM / 32), tDF(DFF / 32, DIM / 32), tFD(DIM / 32, DFF / 32);
  cvt_transpose_kernel<<<tDD, 256, 0, stream>>>(wq,  wqT,  DIM, DIM);
  cvt_transpose_kernel<<<tDD, 256, 0, stream>>>(wk,  wkT,  DIM, DIM);
  cvt_transpose_kernel<<<tDD, 256, 0, stream>>>(wv,  wvT,  DIM, DIM);
  cvt_transpose_kernel<<<tDD, 256, 0, stream>>>(wo1, wo1T, DIM, DIM);
  cvt_transpose_kernel<<<tDD, 256, 0, stream>>>(wo2, wo2T, DIM, DIM);
  cvt_transpose_kernel<<<tDF, 256, 0, stream>>>(fw1, fw1T, DIM, DFF);
  cvt_transpose_kernel<<<tFD, 256, 0, stream>>>(fw2, fw2T, DFF, DIM);

  // 2) h = LN1(x) (f32 residual copy + bf16 matmul copy).
  layernorm_kernel<<<BS, 256, 0, stream>>>(x, ln1s, ln1b, h_f, h_b, DIM);

  // 3) Q/K/V projections (bf16 out).
  dim3 gD(DIM / GBN, BS / GBM);
  gemm_bf16_kernel<<<gD, 256, 0, stream>>>(h_b, wqT, BS, DIM, DIM,
                                           nullptr, nullptr, 0, nullptr, q_b);
  gemm_bf16_kernel<<<gD, 256, 0, stream>>>(h_b, wkT, BS, DIM, DIM,
                                           nullptr, nullptr, 0, nullptr, k_b);
  gemm_bf16_kernel<<<gD, 256, 0, stream>>>(h_b, wvT, BS, DIM, DIM,
                                           nullptr, nullptr, 0, nullptr, v_b);

  // 4) Fused causal attention (no S x S materialization).
  dim3 gA(SEQ / 64, NH, BB);
  flash_attn_kernel<<<gA, 128, 0, stream>>>(q_b, k_b, v_b, at_b);

  // 5) o1 = attn * wo1 ; x1 = h + o1 * wo2.
  gemm_bf16_kernel<<<gD, 256, 0, stream>>>(at_b, wo1T, BS, DIM, DIM,
                                           nullptr, nullptr, 0, nullptr, o1_b);
  gemm_bf16_kernel<<<gD, 256, 0, stream>>>(o1_b, wo2T, BS, DIM, DIM,
                                           nullptr, h_f, 0, x1_f, nullptr);

  // 6) h2 = LN2(x1).
  layernorm_kernel<<<BS, 256, 0, stream>>>(x1_f, ln2s, ln2b, h2_f, h2_b, DIM);

  // 7) FFN: f1 = relu(h2*fw1 + fb1) ; out = h2 + f1*fw2 + fb2.
  dim3 gF1(DFF / GBN, BS / GBM);
  gemm_bf16_kernel<<<gF1, 256, 0, stream>>>(h2_b, fw1T, BS, DFF, DIM,
                                            fb1, nullptr, 1, nullptr, f1_b);
  gemm_bf16_kernel<<<gD, 256, 0, stream>>>(f1_b, fw2T, BS, DIM, DFF,
                                           fb2, h2_f, 0, out, nullptr);
}
